// VectorQuantizer_326417514848
// MI455X (gfx1250) — matrix operand order repaired and checked by reading, not compiled, not run on hardware
//
#include <hip/hip_runtime.h>
#include <hip/hip_bf16.h>

typedef __attribute__((ext_vector_type(16))) _Float16 v16h;
typedef __attribute__((ext_vector_type(8)))  float    v8f;

#define VQ_N 131072
#define VQ_D 64
#define VQ_K 1024
#define CHUNK 256            // codes staged in LDS at a time
#define CSTRIDE 72           // padded f16 elements per code row (144B, 16B aligned)
#define ROWS_PER_BLOCK 128   // 8 waves * 16 rows
#define NTILES (CHUNK / 16)  // 16 code tiles per chunk

__global__ __launch_bounds__(256)
void vq_main(const float* __restrict__ z, const float* __restrict__ emb,
             float* __restrict__ out, float* __restrict__ partial) {
  __shared__ _Float16 lcode[CHUNK * CSTRIDE];
  __shared__ float    lnrm[CHUNK];
  __shared__ float    lpart[8];

  const int tid  = threadIdx.x;
  const int wave = tid >> 5;
  const int lane = tid & 31;
  const int hh   = lane >> 4;     // lane half (0/1)
  const int l15  = lane & 15;
  const int row0 = blockIdx.x * ROWS_PER_BLOCK + wave * 16;

  // ---- A fragments: 16 z-rows quantized to f16, held in VGPRs all kernel ----
  // A k order used here (contiguous, B matches): half h -> K-base 16h; element e -> K = base + e
  v16h a0, a1;
  {
    const float* zr = z + (size_t)(row0 + l15) * VQ_D;
    const int kb = hh * 16;
#pragma unroll
    for (int e = 0; e < 16; ++e) {
      const int k = kb + e;
      a0[e] = (_Float16)zr[k];        // K = 0..31
      a1[e] = (_Float16)zr[k + 32];   // K = 32..63
    }
  }

  float bestv[8];
  int   besti[8];
#pragma unroll
  for (int m = 0; m < 8; ++m) { bestv[m] = -3.402823466e38f; besti[m] = 0; }

  // B-fragment loader: 32x16 f16 tile, column = code; lanes 0-15 K=0..15,
  // lanes 16-31 K=16..31 (plus the per-code 0.5*||e||^2 scalar).
  auto loadB = [&](int t, v16h& b0o, v16h& b1o, float& hno) {
    const _Float16* cb = &lcode[(t * 16 + l15) * CSTRIDE + hh * 16];
#pragma unroll
    for (int e = 0; e < 16; ++e) { b0o[e] = cb[e]; b1o[e] = cb[e + 32]; }
    hno = lnrm[t * 16 + l15];
  };

  for (int ch = 0; ch < VQ_K / CHUNK; ++ch) {
    __syncthreads();                       // previous chunk fully consumed
    const float* src = emb + (size_t)ch * CHUNK * VQ_D;
#pragma unroll
    for (int i = tid; i < CHUNK * VQ_D; i += 256)
      lcode[(i >> 6) * CSTRIDE + (i & 63)] = (_Float16)src[i];
    if (ch + 1 < VQ_K / CHUNK) {           // warm L2/WGP$ for next chunk
      const char* nxt = (const char*)(emb + (size_t)(ch + 1) * CHUNK * VQ_D);
      __builtin_prefetch(nxt + tid * 256, 0, 1);
    }
    __syncthreads();
    {   // 0.5*||e||^2 per code (consistent with the f16-quantized dot path)
      const _Float16* cr = &lcode[tid * CSTRIDE];
      float s = 0.f;
#pragma unroll
      for (int d = 0; d < VQ_D; ++d) { const float v = (float)cr[d]; s += v * v; }
      lnrm[tid] = 0.5f * s;
    }
    __syncthreads();

    // Software-pipelined tile loop: B fragments for tile t+1 are issued
    // before the WMMAs of tile t, so s_wait_dscnt covers a full tile of
    // tensor+VALU work instead of stalling at the consumer.
    v16h b0, b1;
    float hn;
    loadB(0, b0, b1, hn);
#pragma unroll
    for (int t = 0; t < NTILES; ++t) {
      const v16h  c0 = b0, c1 = b1;
      const float chn = hn;
      if (t + 1 < NTILES) loadB(t + 1, b0, b1, hn);

      v8f acc = {};
      acc = __builtin_amdgcn_wmma_f32_16x16x32_f16(false, a0, false, c0,
                                                   (short)0, acc, false, false);
      acc = __builtin_amdgcn_wmma_f32_16x16x32_f16(false, a1, false, c1,
                                                   (short)0, acc, false, false);

      const int code = ch * CHUNK + t * 16 + l15;
#pragma unroll
      for (int m = 0; m < 8; ++m) {        // D layout: lane -> col l15, row m+8*hh
        const float v = acc[m] - chn;      // argmax of (z.e - 0.5||e||^2)
        if (v > bestv[m]) { bestv[m] = v; besti[m] = code; }
      }
    }
  }

  // ---- cross-lane argmax over the 16 columns in each lane-half ----
#pragma unroll
  for (int off = 1; off < 16; off <<= 1) {
#pragma unroll
    for (int m = 0; m < 8; ++m) {
      const float ov = __shfl_xor(bestv[m], off, 32);
      const int   oi = __shfl_xor(besti[m], off, 32);
      if (ov > bestv[m] || (ov == bestv[m] && oi < besti[m])) {
        bestv[m] = ov; besti[m] = oi;
      }
    }
  }

  // ---- gather exact f32 codes, write z_q_st + indices, accumulate loss ----
  float lsum = 0.f;
#pragma unroll
  for (int mm = 0; mm < 16; ++mm) {
    const int idx = __shfl(besti[mm & 7], (mm >> 3) * 16, 32);  // rows 0-7 lane0, 8-15 lane16
    const int row = row0 + mm;
    const float* er = emb + (size_t)idx * VQ_D;
    const float* zr = z   + (size_t)row * VQ_D;
    float* orow     = out + (size_t)row * VQ_D;
    const float e0 = er[lane],      e1 = er[lane + 32];
    const float d0 = e0 - zr[lane], d1 = e1 - zr[lane + 32];
    orow[lane]      = e0;
    orow[lane + 32] = e1;
    lsum += d0 * d0 + d1 * d1;
    if (lane == 0) out[(size_t)VQ_N * VQ_D + row] = (float)idx;
  }
#pragma unroll
  for (int off = 1; off < 32; off <<= 1) lsum += __shfl_xor(lsum, off, 32);
  if (lane == 0) lpart[wave] = lsum;
  __syncthreads();
  if (tid == 0) {
    float s = 0.f;
#pragma unroll
    for (int w = 0; w < 8; ++w) s += lpart[w];
    partial[blockIdx.x] = s;
  }
}

__global__ __launch_bounds__(256)
void vq_loss_reduce(const float* __restrict__ partial, float* __restrict__ out,
                    int nblocks) {
  __shared__ float sh[256];
  const int tid = threadIdx.x;
  float s = 0.f;
  for (int i = tid; i < nblocks; i += 256) s += partial[i];   // fixed order
  sh[tid] = s;
  __syncthreads();
  for (int off = 128; off > 0; off >>= 1) {
    if (tid < off) sh[tid] += sh[tid + off];
    __syncthreads();
  }
  if (tid == 0)
    out[(size_t)VQ_N * VQ_D + VQ_N] =
        sh[0] * 1.25f / (float)((size_t)VQ_N * VQ_D);  // (1+beta)*mean
}

extern "C" void kernel_launch(void* const* d_in, const int* in_sizes, int n_in,
                              void* d_out, int out_size, void* d_ws, size_t ws_size,
                              hipStream_t stream) {
  (void)in_sizes; (void)n_in; (void)out_size; (void)ws_size;
  const float* z   = (const float*)d_in[0];   // [N,1,64] f32
  const float* emb = (const float*)d_in[1];   // [K,64]  f32
  float* out       = (float*)d_out;           // N*64 z_q | N idx | 1 loss
  float* partial   = (float*)d_ws;            // 1024 block partials
  const int nblocks = VQ_N / ROWS_PER_BLOCK;  // 1024
  vq_main<<<nblocks, 256, 0, stream>>>(z, emb, out, partial);
  vq_loss_reduce<<<1, 256, 0, stream>>>(partial, out, nblocks);
}
